// GCL_12610023981467
// MI455X (gfx1250) — matrix operand order repaired and checked by reading, not compiled, not run on hardware
//
#include <hip/hip_runtime.h>
#include <hip/hip_bf16.h>
#include <stdint.h>

#define IN_DIM   128
#define OUT_DIM  128
#define EDGE_DIM 16
#define N_NODES  20000
#define N_EDGES  640000
#define EPB      64        // edges per block in edge kernel

typedef __attribute__((ext_vector_type(16))) __bf16 bf16x16;
typedef __attribute__((ext_vector_type(8)))  float  f32x8;

union FragB16 { uint32_t u[8]; bf16x16 v; };

__device__ __forceinline__ uint16_t f2bf(float f) {
  uint32_t u = __float_as_uint(f);
  u += 0x7FFFu + ((u >> 16) & 1u);          // round-to-nearest-even
  return (uint16_t)(u >> 16);
}

__device__ __forceinline__ float silu_f(float x) {
  return x / (1.0f + __expf(-x));
}

__device__ __forceinline__ f32x8 wmma_bf16(const FragB16& a, const FragB16& b, f32x8 c) {
  return __builtin_amdgcn_wmma_f32_16x16x32_bf16(false, a.v, false, b.v,
                                                 (short)0, c, false, false);
}

// A-fragment (16-bit, 16x32): lane = row m (mod 16); klo = 8*(lane>=16);
// VGPR i in 0..3 -> K = kb+klo+2i,+1 ; VGPR 4+i -> K = kb+klo+16+2i,+1
template <int STRIDE>
__device__ __forceinline__ void load_afrag(const uint16_t (*M)[STRIDE], int m, int kb,
                                           FragB16& a) {
#pragma unroll
  for (int i = 0; i < 4; ++i) a.u[i]     = *(const uint32_t*)&M[m][kb + 2 * i];
#pragma unroll
  for (int i = 0; i < 4; ++i) a.u[4 + i] = *(const uint32_t*)&M[m][kb + 16 + 2 * i];
}

// B-fragment load from packed weights: 8 consecutive dwords per lane.
__device__ __forceinline__ void load_bfrag(const uint32_t* __restrict__ Wp, int kt, int nt,
                                           int lane, FragB16& b) {
  const uint4* p = (const uint4*)(Wp + ((size_t)((kt * 8 + nt) * 32 + lane) * 8));
  uint4 lo = p[0], hi = p[1];
  b.u[0] = lo.x; b.u[1] = lo.y; b.u[2] = lo.z; b.u[3] = lo.w;
  b.u[4] = hi.x; b.u[5] = hi.y; b.u[6] = hi.z; b.u[7] = hi.w;
}

__global__ void zero_kernel(float* __restrict__ p, int n) {
  int i = blockIdx.x * blockDim.x + threadIdx.x;
  if (i < n) p[i] = 0.0f;
}

// Repack W (Kact x 128, f32 row-major) into bf16 WMMA B-fragment order.
// dword index = ((kt*8 + nt)*32 + lane)*8 + i
// B layout: col n = nt*16 + (lane&15); K = kt*32 + (lane>>4)*8 + (i>>2)*16 + 2*(i&3) (pair K,K+1)
__global__ void pack_kernel(const float* __restrict__ W, int Kact,
                            uint32_t* __restrict__ out, int total) {
  int tid = blockIdx.x * blockDim.x + threadIdx.x;
  if (tid >= total) return;
  int i    = tid & 7;
  int lane = (tid >> 3) & 31;
  int nt   = (tid >> 8) & 7;
  int kt   = tid >> 11;
  int n = nt * 16 + (lane & 15);
  int k = kt * 32 + (lane >> 4) * 8 + (i >> 2) * 16 + 2 * (i & 3);
  float a = (k     < Kact) ? W[(size_t)k * 128 + n] : 0.0f;
  float b = (k + 1 < Kact) ? W[(size_t)(k + 1) * 128 + n] : 0.0f;
  out[tid] = (uint32_t)f2bf(a) | ((uint32_t)f2bf(b) << 16);
}

// ---- Precompute P[n][0:128] = h[n]@W_m1[0:128,:], P[n][128:256] = h[n]@W_m1[128:256,:]
// 16 nodes per block, 16 wave32s (512 threads), one N-tile per wave.
__global__ void __launch_bounds__(512)
precomp_kernel(const float* __restrict__ h, const uint32_t* __restrict__ Wm1rP,
               const uint32_t* __restrict__ Wm1cP, float* __restrict__ P) {
  __shared__ __align__(16) uint16_t A[16][128];
  const int tid = threadIdx.x;
  const int n0  = blockIdx.x * 16;

  for (int idx = tid; idx < 16 * 128; idx += 512) {
    int n = idx >> 7, c = idx & 127;
    A[n][c] = f2bf(h[(size_t)(n0 + n) * 128 + c]);
  }
  __syncthreads();

  const int lane = tid & 31;
  const int w    = tid >> 5;          // 0..15
  const int m    = lane & 15;
  const int hi   = lane >> 4;
  const int klo  = hi * 8;
  const int nt   = w & 7;
  const uint32_t* Wp = (w < 8) ? Wm1rP : Wm1cP;
  const int cbase = (w < 8 ? 0 : 128) + nt * 16 + m;

  f32x8 acc = {0.f, 0.f, 0.f, 0.f, 0.f, 0.f, 0.f, 0.f};
#pragma unroll
  for (int kt = 0; kt < 4; ++kt) {
    FragB16 a, b;
    load_afrag<128>(A, m, kt * 32 + klo, a);
    load_bfrag(Wp, kt, nt, lane, b);
    acc = wmma_bf16(a, b, acc);
  }
#pragma unroll
  for (int v = 0; v < 8; ++v)
    P[(size_t)(n0 + v + 8 * hi) * 256 + cbase] = acc[v];
}

// ---------------- Edge kernel: 64 edges per block, 8 wave32s ----------------
// msg1 = P1[row] + P2[col] + ea@W_m1[256:272] + b_m1  (gather + 1 WMMA per subtile)
// B-fragments for both GEMMs held in registers and reused across 4 M-subtiles.
__global__ void __launch_bounds__(256)
edge_kernel(const float* __restrict__ P, const int* __restrict__ ei,
            const float* __restrict__ ea, const float* __restrict__ emask,
            const uint32_t* __restrict__ WeaP, const float* __restrict__ bm1,
            const uint32_t* __restrict__ Wm2p, const float* __restrict__ bm2,
            const float* __restrict__ Wa, const float* __restrict__ ba,
            float* __restrict__ agg) {
  __shared__ __align__(16) uint16_t eaA[EPB][32];    // edge_attr bf16, K pad 16->32
  __shared__ __align__(16) uint16_t msgB[EPB][128];  // hidden after layer1+silu (bf16)
  __shared__ __align__(16) float    msg2[EPB][128];  // after GEMM2+silu (f32)
  __shared__ float part[EPB][4];
  __shared__ float atts[EPB];
  __shared__ int   rowi[EPB], coli[EPB];

  const int tid = threadIdx.x;
  const int e0  = blockIdx.x * EPB;

  if (tid < EPB) {
    rowi[tid] = ei[e0 + tid];
    coli[tid] = ei[N_EDGES + e0 + tid];
  }

  // Build edge_attr tile: [EPB][32] bf16 (zero padded)
  for (int idx = tid; idx < EPB * 32; idx += 256) {
    int e = idx >> 5, c = idx & 31;
    eaA[e][c] = (c < EDGE_DIM)
                  ? f2bf(ea[(size_t)(e0 + e) * EDGE_DIM + c]) : (uint16_t)0;
  }
  __syncthreads();

  const int lane = tid & 31;
  const int w    = tid >> 5;          // N-tile 0..7
  const int m    = lane & 15;
  const int hi   = lane >> 4;
  const int klo  = hi * 8;
  const int colg = w * 16 + m;

  // Persistent B-fragments (reused across all 4 M-subtiles)
  FragB16 bea;
  load_bfrag(WeaP, 0, w, lane, bea);
  FragB16 bw[4];
#pragma unroll
  for (int kt = 0; kt < 4; ++kt) load_bfrag(Wm2p, kt, w, lane, bw[kt]);

  const float bias1 = bm1[colg];
  const float bias2 = bm2[colg];

  // Layer 1 over 4 subtiles: WMMA on edge_attr + gathered node terms -> msgB
#pragma unroll
  for (int t = 0; t < 4; ++t) {
    FragB16 a;
    load_afrag<32>((const uint16_t (*)[32])&eaA[t * 16][0], m, klo, a);
    f32x8 acc = {0.f, 0.f, 0.f, 0.f, 0.f, 0.f, 0.f, 0.f};
    acc = wmma_bf16(a, bea, acc);
#pragma unroll
    for (int v = 0; v < 8; ++v) {
      int mv = t * 16 + v + 8 * hi;
      float s = acc[v] + bias1
              + P[(size_t)rowi[mv] * 256 + colg]
              + P[(size_t)coli[mv] * 256 + 128 + colg];
      msgB[mv][colg] = f2bf(silu_f(s));
    }
  }
  __syncthreads();

  // GEMM2 over 4 subtiles: [16x128] @ [128x128], 4 WMMA steps each
#pragma unroll
  for (int t = 0; t < 4; ++t) {
    f32x8 acc2 = {0.f, 0.f, 0.f, 0.f, 0.f, 0.f, 0.f, 0.f};
#pragma unroll
    for (int kt = 0; kt < 4; ++kt) {
      FragB16 a;
      load_afrag<128>((const uint16_t (*)[128])&msgB[t * 16][0], m, kt * 32 + klo, a);
      acc2 = wmma_bf16(a, bw[kt], acc2);
    }
#pragma unroll
    for (int v = 0; v < 8; ++v)
      msg2[t * 16 + v + 8 * hi][colg] = silu_f(acc2[v] + bias2);
  }
  __syncthreads();

  // Attention: att[e] = sigmoid(msg2[e,:] . Wa + ba) * edge_mask[e]
  {
    int r = tid >> 2, seg = tid & 3;        // 64 rows x 4 segments of 32 cols
    float s = 0.0f;
#pragma unroll
    for (int j = 0; j < 32; ++j) {
      int c = seg * 32 + j;
      s += msg2[r][c] * Wa[c];
    }
    part[r][seg] = s;
  }
  __syncthreads();
  if (tid < EPB) {
    float s = ba[0] + part[tid][0] + part[tid][1] + part[tid][2] + part[tid][3];
    atts[tid] = (1.0f / (1.0f + __expf(-s))) * emask[e0 + tid];
  }
  __syncthreads();

  // Scatter-add into agg[row]
  for (int idx = tid; idx < EPB * 128; idx += 256) {
    int e = idx >> 7, c = idx & 127;
    float val = msg2[e][c] * atts[e];
    __hip_atomic_fetch_add(&agg[(size_t)rowi[e] * 128 + c], val,
                           __ATOMIC_RELAXED, __HIP_MEMORY_SCOPE_AGENT);
  }
}

// ---------------- Node kernel: 16 nodes per block, 8 wave32s ----------------
__global__ void __launch_bounds__(256)
node_kernel(const float* __restrict__ h, const float* __restrict__ agg,
            const uint32_t* __restrict__ Wn1p, const float* __restrict__ bn1,
            const uint32_t* __restrict__ Wn2p, const float* __restrict__ bn2,
            const float* __restrict__ flags, float* __restrict__ out) {
  __shared__ __align__(16) uint16_t A[16][256];     // concat(h, agg) bf16
  __shared__ __align__(16) uint16_t hid[16][128];   // after GEMM1+silu

  const int tid = threadIdx.x;
  const int n0  = blockIdx.x * 16;

  for (int idx = tid; idx < 16 * 128; idx += 256) {
    int n = idx >> 7, c = idx & 127;
    A[n][c]       = f2bf(h[(size_t)(n0 + n) * 128 + c]);
    A[n][128 + c] = f2bf(agg[(size_t)(n0 + n) * 128 + c]);
  }
  __syncthreads();

  const int lane = tid & 31;
  const int w    = tid >> 5;
  const int m    = lane & 15;
  const int hi   = lane >> 4;
  const int klo  = hi * 8;
  const int colg = w * 16 + m;

  // GEMM1: [16x256] @ [256x128], 8 WMMA steps
  f32x8 acc = {0.f, 0.f, 0.f, 0.f, 0.f, 0.f, 0.f, 0.f};
#pragma unroll
  for (int kt = 0; kt < 8; ++kt) {
    FragB16 a, b;
    load_afrag<256>(A, m, kt * 32 + klo, a);
    load_bfrag(Wn1p, kt, w, lane, b);
    acc = wmma_bf16(a, b, acc);
  }
  {
    float bias = bn1[colg];
#pragma unroll
    for (int v = 0; v < 8; ++v)
      hid[v + 8 * hi][colg] = f2bf(silu_f(acc[v] + bias));
  }
  __syncthreads();

  // GEMM2: [16x128] @ [128x128], 4 WMMA steps (no activation, bias only)
  f32x8 acc2 = {0.f, 0.f, 0.f, 0.f, 0.f, 0.f, 0.f, 0.f};
#pragma unroll
  for (int kt = 0; kt < 4; ++kt) {
    FragB16 a, b;
    load_afrag<128>(hid, m, kt * 32 + klo, a);
    load_bfrag(Wn2p, kt, w, lane, b);
    acc2 = wmma_bf16(a, b, acc2);
  }
  {
    float bias = bn2[colg];
#pragma unroll
    for (int v = 0; v < 8; ++v) {
      int n = n0 + v + 8 * hi;
      float upd = acc2[v] + bias;
      out[(size_t)n * 128 + colg] =
          (h[(size_t)n * 128 + colg] + upd) * flags[n];
    }
  }
}

extern "C" void kernel_launch(void* const* d_in, const int* in_sizes, int n_in,
                              void* d_out, int out_size, void* d_ws, size_t ws_size,
                              hipStream_t stream) {
  (void)in_sizes; (void)n_in; (void)out_size; (void)ws_size;
  const float* h     = (const float*)d_in[0];
  const int*   ei    = (const int*)  d_in[1];
  const float* ea    = (const float*)d_in[2];
  const float* flags = (const float*)d_in[3];
  const float* emask = (const float*)d_in[4];
  const float* Wm1   = (const float*)d_in[5];
  const float* bm1   = (const float*)d_in[6];
  const float* Wm2   = (const float*)d_in[7];
  const float* bm2   = (const float*)d_in[8];
  const float* Wa    = (const float*)d_in[9];
  const float* ba    = (const float*)d_in[10];
  const float* Wn1   = (const float*)d_in[11];
  const float* bn1   = (const float*)d_in[12];
  const float* Wn2   = (const float*)d_in[13];
  const float* bn2   = (const float*)d_in[14];
  float* out = (float*)d_out;

  // Workspace layout
  char* ws = (char*)d_ws;
  size_t off = 0;
  float* agg = (float*)(ws + off);  off += (size_t)N_NODES * 128 * sizeof(float);
  float* P   = (float*)(ws + off);  off += (size_t)N_NODES * 256 * sizeof(float);
  uint32_t* Wm1rP = (uint32_t*)(ws + off); off += (size_t)8192  * 4;  // W_m1 rows 0..127
  uint32_t* Wm1cP = (uint32_t*)(ws + off); off += (size_t)8192  * 4;  // W_m1 rows 128..255
  uint32_t* WeaP  = (uint32_t*)(ws + off); off += (size_t)2048  * 4;  // W_m1 rows 256..271
  uint32_t* Wm2p  = (uint32_t*)(ws + off); off += (size_t)8192  * 4;
  uint32_t* Wn1p  = (uint32_t*)(ws + off); off += (size_t)16384 * 4;
  uint32_t* Wn2p  = (uint32_t*)(ws + off); off += (size_t)8192  * 4;

  const int aggN = N_NODES * 128;
  zero_kernel<<<(aggN + 255) / 256, 256, 0, stream>>>(agg, aggN);

  pack_kernel<<<(8192  + 255) / 256, 256, 0, stream>>>(Wm1,             128, Wm1rP, 8192);
  pack_kernel<<<(8192  + 255) / 256, 256, 0, stream>>>(Wm1 + 128 * 128, 128, Wm1cP, 8192);
  pack_kernel<<<(2048  + 255) / 256, 256, 0, stream>>>(Wm1 + 256 * 128,  16, WeaP,  2048);
  pack_kernel<<<(8192  + 255) / 256, 256, 0, stream>>>(Wm2,             128, Wm2p,  8192);
  pack_kernel<<<(16384 + 255) / 256, 256, 0, stream>>>(Wn1,             256, Wn1p, 16384);
  pack_kernel<<<(8192  + 255) / 256, 256, 0, stream>>>(Wn2,             128, Wn2p,  8192);

  precomp_kernel<<<N_NODES / 16, 512, 0, stream>>>(h, Wm1rP, Wm1cP, P);

  edge_kernel<<<N_EDGES / EPB, 256, 0, stream>>>(P, ei, ea, emask,
                                                 WeaP, bm1, Wm2p, bm2, Wa, ba, agg);

  node_kernel<<<N_NODES / 16, 256, 0, stream>>>(h, agg, Wn1p, bn1, Wn2p, bn2,
                                                flags, out);
}
